// GIN_regression_87282325390050
// MI455X (gfx1250) — matrix operand order repaired and checked by reading, not compiled, not run on hardware
//
#include <hip/hip_runtime.h>

typedef __bf16 bf16;
typedef __attribute__((ext_vector_type(16))) bf16  v16bf;
typedef __attribute__((ext_vector_type(8)))  bf16  v8bf;
typedef __attribute__((ext_vector_type(4)))  bf16  v4bf;
typedef __attribute__((ext_vector_type(8)))  float v8f;

#define LEAKY_ALPHA 0.2f

// ---------------------------------------------------------------- utilities

__global__ __launch_bounds__(256) void k_copy_f32v4(const float4* __restrict__ in,
                                                    float4* __restrict__ out, int n4) {
  int i = blockIdx.x * 256 + threadIdx.x;
  if (i < n4) out[i] = in[i];
}

__global__ __launch_bounds__(256) void k_cast_f32_bf16v4(const float4* __restrict__ in,
                                                         v4bf* __restrict__ out, int n4) {
  int i = blockIdx.x * 256 + threadIdx.x;
  if (i >= n4) return;
  float4 v = in[i];
  v4bf o = { (bf16)v.x, (bf16)v.y, (bf16)v.z, (bf16)v.w };
  out[i] = o;
}

// W [K,F] f32 row-major  ->  WT [F,K] bf16 row-major
__global__ __launch_bounds__(256) void k_transpose_cast(const float* __restrict__ W,
                                                        bf16* __restrict__ WT,
                                                        int K, int F) {
  int i = blockIdx.x * 256 + threadIdx.x;
  if (i >= K * F) return;
  int k = i / F, f = i % F;
  WT[(size_t)f * K + k] = (bf16)W[i];
}

// ---------------------------------------------------------------- scatter-add
// One wave per edge; lane handles F/32 consecutive features. Relaxed device-
// scope non-returning f32 atomics -> global_atomic_add_u32/f32 into L2.

template <int F>
__global__ __launch_bounds__(256) void k_scatter_add(const float* __restrict__ x,
                                                     const int* __restrict__ src,
                                                     const int* __restrict__ dst,
                                                     float* __restrict__ agg, int E) {
  int g = blockIdx.x * 256 + threadIdx.x;
  int e = g >> 5, lane = g & 31;
  if (e >= E) return;
  const int s = src[e], d = dst[e];
  constexpr int C = F / 32;  // 4 or 8 floats per lane
  const float* xs = x + (size_t)s * F + lane * C;
  float* ad = agg + (size_t)d * F + lane * C;
#pragma unroll
  for (int i = 0; i < C; ++i)
    (void)__hip_atomic_fetch_add(&ad[i], xs[i], __ATOMIC_RELAXED,
                                 __HIP_MEMORY_SCOPE_AGENT);
}

// ---------------------------------------------------------------- WMMA GEMM
// C[M,F] = leaky(A[M,K] @ W[K,F] + b), with W given pre-transposed as
// WT[F,K] bf16. One wave computes a 16x64 output strip (4 accumulators).
// A layout per lane (m=l&15, half=l>>4): elems 0..7 -> K=k0+8*half..,
//                                        elems 8..15 -> K=k0+16+8*half..
// B layout per lane (n=l&15, half=l>>4): same as A (elems 0..7 -> K=k0+8*half..,
//                                        elems 8..15 -> K=k0+16+8*half..)

template <int K, int F, bool OUT_BF16>
__global__ __launch_bounds__(256) void k_gemm_bias_leaky(
    const bf16* __restrict__ A, const bf16* __restrict__ WT,
    const float* __restrict__ bias, bf16* __restrict__ outb,
    float* __restrict__ outf, int M) {
  constexpr int NT = 4;                    // 4 x 16 = 64 output columns per wave
  constexpr int wavesPerM = F / (16 * NT); // 4 (F=256) or 2 (F=128)
  const int wave = (blockIdx.x * 256 + threadIdx.x) >> 5;
  const int lane = threadIdx.x & 31;
  const int mTile = wave / wavesPerM;
  const int nGrp  = wave % wavesPerM;
  if (mTile * 16 >= M) return;             // wave-uniform: EXEC stays all-ones

  const int mlo  = lane & 15;
  const int half = lane >> 4;
  const int row  = mTile * 16 + mlo;
  const int col0 = nGrp * (16 * NT);

  v8f acc[NT] = {};

  const bf16* aRow = A + (size_t)row * K + half * 8;
  const bf16* bCol = WT + (size_t)(col0 + mlo) * K + half * 8;

#pragma unroll
  for (int k0 = 0; k0 < K; k0 += 32) {
    if (k0 + 32 < K) __builtin_prefetch(aRow + k0 + 32, 0, 3);
    v8bf alo = *(const v8bf*)(aRow + k0);
    v8bf ahi = *(const v8bf*)(aRow + k0 + 16);
    v16bf a;
#pragma unroll
    for (int i = 0; i < 8; ++i) { a[i] = alo[i]; a[8 + i] = ahi[i]; }

#pragma unroll
    for (int t = 0; t < NT; ++t) {
      const bf16* bRow = bCol + (size_t)(t * 16) * K + k0;
      v8bf blo = *(const v8bf*)(bRow);
      v8bf bhi = *(const v8bf*)(bRow + 16);
      v16bf b;
#pragma unroll
      for (int i = 0; i < 8; ++i) { b[i] = blo[i]; b[8 + i] = bhi[i]; }
      acc[t] = __builtin_amdgcn_wmma_f32_16x16x32_bf16(
          /*neg_a=*/false, a, /*neg_b=*/false, b,
          /*c_mod=*/(short)0, acc[t], /*reuse_a=*/false, /*reuse_b=*/false);
    }
  }

  // Epilogue: bias + leaky-relu, scatter 8 rows per accumulator.
#pragma unroll
  for (int t = 0; t < NT; ++t) {
    const int n = col0 + t * 16 + mlo;
    const float bv = bias[n];
#pragma unroll
    for (int r = 0; r < 8; ++r) {
      const int m = mTile * 16 + r + half * 8;
      float v = acc[t][r] + bv;
      v = (v >= 0.0f) ? v : LEAKY_ALPHA * v;
      if (OUT_BF16) outb[(size_t)m * F + n] = (bf16)v;
      else          outf[(size_t)m * F + n] = v;
    }
  }
}

// ---------------------------------------------------------------- final linear
// out[i] = h2[i,:128] . W5 + b5 ; one wave per row, shfl reduction.

__global__ __launch_bounds__(256) void k_final_dot(const float* __restrict__ h,
                                                   const float* __restrict__ W5,
                                                   const float* __restrict__ b5,
                                                   float* __restrict__ out, int M) {
  int wave = (blockIdx.x * 256 + threadIdx.x) >> 5;
  int lane = threadIdx.x & 31;
  if (wave >= M) return;
  float4 hv = *(const float4*)(h + (size_t)wave * 128 + lane * 4);
  float4 wv = *(const float4*)(W5 + lane * 4);
  float s = hv.x * wv.x + hv.y * wv.y + hv.z * wv.z + hv.w * wv.w;
#pragma unroll
  for (int off = 16; off > 0; off >>= 1) s += __shfl_down(s, off, 32);
  if (lane == 0) out[wave] = s + b5[0];
}

// ---------------------------------------------------------------- launch

template <int K, int F, bool OUT_BF16>
static void launch_gemm(const bf16* A, const bf16* WT, const float* bias,
                        bf16* outb, float* outf, int M, hipStream_t stream) {
  constexpr int NT = 4;
  int waves = (M / 16) * (F / (16 * NT));
  int blocks = (waves * 32 + 255) / 256;
  k_gemm_bias_leaky<K, F, OUT_BF16><<<blocks, 256, 0, stream>>>(A, WT, bias, outb, outf, M);
}

extern "C" void kernel_launch(void* const* d_in, const int* in_sizes, int n_in,
                              void* d_out, int out_size, void* d_ws, size_t ws_size,
                              hipStream_t stream) {
  const float* x  = (const float*)d_in[0];
  const int*   ei = (const int*)d_in[2];
  const float* W1 = (const float*)d_in[3];  const float* b1 = (const float*)d_in[4];
  const float* W2 = (const float*)d_in[5];  const float* b2 = (const float*)d_in[6];
  const float* W3 = (const float*)d_in[7];  const float* b3 = (const float*)d_in[8];
  const float* W4 = (const float*)d_in[9];  const float* b4 = (const float*)d_in[10];
  const float* W5 = (const float*)d_in[11]; const float* b5 = (const float*)d_in[12];
  float* out = (float*)d_out;

  const int N = in_sizes[0] / 128;   // 50000 (multiple of 16)
  const int E = in_sizes[2] / 2;     // 800000
  const int* src = ei;
  const int* dst = ei + E;

  // Workspace layout (all offsets 256B aligned):
  char* ws = (char*)d_ws;
  const size_t SZ_F32 = (size_t)N * 256 * sizeof(float);  // 51.2 MB
  const size_t SZ_B16 = (size_t)N * 256 * sizeof(bf16);   // 25.6 MB
  float* S0 = (float*)(ws);                   // agg1 [N,128] -> agg2 [N,256]
  float* S1 = (float*)(ws + SZ_F32);          // h1 [N,256]  -> h2 [N,128]
  bf16*  S2 = (bf16*) (ws + 2 * SZ_F32);                // a1 -> a2 (bf16)
  bf16*  S3 = (bf16*) (ws + 2 * SZ_F32 + SZ_B16);       // z1 -> z3 (bf16)
  bf16*  W1T = (bf16*)(ws + 2 * SZ_F32 + 2 * SZ_B16);   // [256,128]
  bf16*  W2T = W1T + 256 * 128;                         // [256,256]
  bf16*  W3T = W2T + 256 * 256;                         // [256,256]
  bf16*  W4T = W3T + 256 * 256;                         // [128,256]

  // Weight transposes + bf16 casts (tiny, once per launch)
  k_transpose_cast<<<(128 * 256 + 255) / 256, 256, 0, stream>>>(W1, W1T, 128, 256);
  k_transpose_cast<<<(256 * 256 + 255) / 256, 256, 0, stream>>>(W2, W2T, 256, 256);
  k_transpose_cast<<<(256 * 256 + 255) / 256, 256, 0, stream>>>(W3, W3T, 256, 256);
  k_transpose_cast<<<(256 * 128 + 255) / 256, 256, 0, stream>>>(W4, W4T, 256, 128);

  // ---- conv1 ----
  int n4_128 = N * 128 / 4;
  k_copy_f32v4<<<(n4_128 + 255) / 256, 256, 0, stream>>>((const float4*)x, (float4*)S0, n4_128);
  k_scatter_add<128><<<(E * 32 + 255) / 256, 256, 0, stream>>>(x, src, dst, S0, E);
  k_cast_f32_bf16v4<<<(n4_128 + 255) / 256, 256, 0, stream>>>((const float4*)S0, (v4bf*)S2, n4_128);
  launch_gemm<128, 256, true >(S2, W1T, b1, S3, nullptr, N, stream);   // z1 bf16
  launch_gemm<256, 256, false>(S3, W2T, b2, nullptr, S1, N, stream);   // h1 f32

  // ---- conv2 ----
  int n4_256 = N * 256 / 4;
  k_copy_f32v4<<<(n4_256 + 255) / 256, 256, 0, stream>>>((const float4*)S1, (float4*)S0, n4_256);
  k_scatter_add<256><<<(E * 32 + 255) / 256, 256, 0, stream>>>(S1, src, dst, S0, E);
  k_cast_f32_bf16v4<<<(n4_256 + 255) / 256, 256, 0, stream>>>((const float4*)S0, (v4bf*)S2, n4_256);
  launch_gemm<256, 256, true >(S2, W3T, b3, S3, nullptr, N, stream);   // z3 bf16
  launch_gemm<256, 128, false>(S3, W4T, b4, nullptr, S1, N, stream);   // h2 f32

  // ---- final linear [N,128] -> [N,1] ----
  k_final_dot<<<(N * 32 + 255) / 256, 256, 0, stream>>>(S1, W5, b5, out, N);
  (void)n_in; (void)out_size; (void)ws_size;
}
